// PiNoiseBiLoRA_23192823399005
// MI455X (gfx1250) — matrix operand order repaired and checked by reading, not compiled, not run on hardware
//
#include <hip/hip_runtime.h>
#include <hip/hip_bf16.h>

// ---------------------------------------------------------------------------
// PiNoiseBiLoRA on MI455X (gfx1250):
//   out = irfftBasis( MLPs( dftBasis(x) ) )  -- everything as f16 WMMA GEMMs
//   with f32 accumulation. 76-dim padded to 96 (6 N-tiles / 3 K-steps of 32).
//   Fragment loads are software-pipelined so each WMMA group waits on loads
//   issued one full iteration earlier (hides L2 latency inside a wave).
// ---------------------------------------------------------------------------

typedef __attribute__((ext_vector_type(16))) _Float16 v16h;
typedef __attribute__((ext_vector_type(8)))  _Float16 v8h;
typedef __attribute__((ext_vector_type(8)))  float    v8f;

#define IN_DIM 768
#define KSEL   38
#define MLP_IN 76
#define PAD    96
#define HID    256
#define NROWS  32768   // 8 * 4096

// ---- WMMA wrapper (V_WMMA_F32_16X16X32_F16) -------------------------------
__device__ __forceinline__ v8f wmma_f32(v16h a, v16h b, v8f c) {
  return __builtin_amdgcn_wmma_f32_16x16x32_f16(
      /*neg_a=*/false, a, /*neg_b=*/false, b,
      /*c_mod=*/(short)0, c, /*reuse_a=*/false, /*reuse_b=*/false);
}

__device__ __forceinline__ v8f zero8() {
  return (v8f){0.f, 0.f, 0.f, 0.f, 0.f, 0.f, 0.f, 0.f};
}

// ---- Fragment loaders (layouts per ISA 7.12.2, wave32) --------------------
// A: 16x32 f16, row-major source with leading dim lda (halfs).
// lane l: m=l&15, kh=l>>4 ; elem e<8 -> k=8*kh+e ; e>=8 -> k=16+8*kh+(e-8)
__device__ __forceinline__ v16h load_a_f16(const _Float16* base, int lda, int k0) {
  const int l = threadIdx.x & 31, m = l & 15, kh = l >> 4;
  const _Float16* p = base + (size_t)m * lda + k0 + 8 * kh;
  v8h lo = *(const v8h*)p;
  v8h hi = *(const v8h*)(p + 16);
  v16h a;
#pragma unroll
  for (int e = 0; e < 8; ++e) { a[e] = lo[e]; a[e + 8] = hi[e]; }
  return a;
}

// A from f32 source, converting to f16 on the fly.
__device__ __forceinline__ v16h load_a_f32(const float* base, int lda, int k0) {
  const int l = threadIdx.x & 31, m = l & 15, kh = l >> 4;
  const float* p = base + (size_t)m * lda + k0 + 8 * kh;
  v16h a;
#pragma unroll
  for (int e = 0; e < 8; ++e) { a[e] = (_Float16)p[e]; a[e + 8] = (_Float16)p[e + 16]; }
  return a;
}

// B: 32x16 f16 stored N-major, K-contiguous (ldb halfs per N row).
// lane l: n=l&15, kh=l>>4 ; elem e<8 -> k=8*kh+e ; e>=8 -> k=16+8*kh+(e-8)
__device__ __forceinline__ v16h load_b(const _Float16* base, int ldb, int n0, int k0) {
  const int l = threadIdx.x & 31, n = l & 15, kh = l >> 4;
  const _Float16* p = base + (size_t)(n0 + n) * ldb + k0 + 8 * kh;
  v8h lo = *(const v8h*)p;
  v8h hi = *(const v8h*)(p + 16);
  v16h b;
#pragma unroll
  for (int e = 0; e < 8; ++e) { b[e] = lo[e]; b[e + 8] = hi[e]; }
  return b;
}

// ---------------------------------------------------------------------------
// Kernel 0: build f16 bases + padded f16 weights in workspace.
//   FbT [96][768] : forward DFT basis, N-major (N = spectral component)
//   Gm  [768][96] : inverse basis, N-major (N = output sample)
//   W1  [256][96] : w1 (HID x MLP_IN) zero-padded K
//   W2  [96][256] : w2 (MLP_IN x HID) zero-padded N
// ---------------------------------------------------------------------------
__global__ void setup_kernel(const int* __restrict__ idx,
                             const float* __restrict__ mu_w1, const float* __restrict__ mu_w2,
                             const float* __restrict__ sg_w1, const float* __restrict__ sg_w2,
                             _Float16* __restrict__ FbT, _Float16* __restrict__ Gm,
                             _Float16* __restrict__ W1mu, _Float16* __restrict__ W1sg,
                             _Float16* __restrict__ W2mu, _Float16* __restrict__ W2sg) {
  const int tid = blockIdx.x * blockDim.x + threadIdx.x;
  const int nth = gridDim.x * blockDim.x;
  const float w = 6.2831853071795864769f / 768.0f;

  for (int i = tid; i < PAD * IN_DIM; i += nth) {        // FbT[n][j]
    int n = i / IN_DIM, j = i % IN_DIM;
    float v = 0.f;
    if (n < KSEL)            { int ph = (idx[n] * j) % 768;        v =  cosf(w * (float)ph); }
    else if (n < 2 * KSEL)   { int ph = (idx[n - KSEL] * j) % 768; v = -sinf(w * (float)ph); }
    FbT[i] = (_Float16)v;
  }
  const float sc = 2.0f / 768.0f;
  for (int i = tid; i < IN_DIM * PAD; i += nth) {        // Gm[n][k]
    int n = i / PAD, k = i % PAD;
    float v = 0.f;
    if (k < KSEL)            { int ph = (idx[k] * n) % 768;        v =  sc * cosf(w * (float)ph); }
    else if (k < 2 * KSEL)   { int ph = (idx[k - KSEL] * n) % 768; v = -sc * sinf(w * (float)ph); }
    Gm[i] = (_Float16)v;
  }
  for (int i = tid; i < HID * PAD; i += nth) {           // W1[r][k]
    int r = i / PAD, k = i % PAD;
    W1mu[i] = (_Float16)((k < MLP_IN) ? mu_w1[r * MLP_IN + k] : 0.f);
    W1sg[i] = (_Float16)((k < MLP_IN) ? sg_w1[r * MLP_IN + k] : 0.f);
  }
  for (int i = tid; i < PAD * HID; i += nth) {           // W2[n][k]
    int n = i / HID, k = i % HID;
    W2mu[i] = (_Float16)((n < MLP_IN) ? mu_w2[n * HID + k] : 0.f);
    W2sg[i] = (_Float16)((n < MLP_IN) ? sg_w2[n * HID + k] : 0.f);
  }
}

// ---------------------------------------------------------------------------
// Kernel 1: Xin[32768][96] (f16) = x (f32, 32768x768) @ FbT^T
// 4 waves/block, each wave owns 16 rows x all 96 cols; K loop 24 steps of 32.
// Software-pipelined: step ks+1 fragments are in flight during step ks WMMAs.
// ---------------------------------------------------------------------------
__global__ __launch_bounds__(128) void gemm1_kernel(const float* __restrict__ x,
                                                    const _Float16* __restrict__ FbT,
                                                    _Float16* __restrict__ Xin) {
  const int wave = threadIdx.x >> 5;
  const size_t row0 = (size_t)(blockIdx.x * 4 + wave) * 16;
  const float* xr = x + row0 * IN_DIM;

  v8f c[6];
#pragma unroll
  for (int t = 0; t < 6; ++t) c[t] = zero8();

  v16h a = load_a_f32(xr, IN_DIM, 0);
  v16h b[6];
#pragma unroll
  for (int t = 0; t < 6; ++t) b[t] = load_b(FbT, IN_DIM, t * 16, 0);

  for (int ks = 0; ks < 24; ++ks) {
    const int kn = (ks < 23) ? (ks + 1) * 32 : 0;   // last-iter reload is harmless
    v16h an = load_a_f32(xr, IN_DIM, kn);
    v16h bn[6];
#pragma unroll
    for (int t = 0; t < 6; ++t) bn[t] = load_b(FbT, IN_DIM, t * 16, kn);
#pragma unroll
    for (int t = 0; t < 6; ++t) c[t] = wmma_f32(a, b[t], c[t]);
    a = an;
#pragma unroll
    for (int t = 0; t < 6; ++t) b[t] = bn[t];
  }

  const int l = threadIdx.x & 31, n = l & 15, kh = l >> 4;
#pragma unroll
  for (int t = 0; t < 6; ++t)
#pragma unroll
    for (int e = 0; e < 8; ++e)
      Xin[(row0 + e + 8 * kh) * PAD + t * 16 + n] = (_Float16)c[t][e];
}

// ---------------------------------------------------------------------------
// Kernel 2: fused dual MLP + reparameterization. 1 wave / block, 16 rows.
//   h = Xin @ W1^T + b1 ; LayerNorm ; exact GELU ; o = h @ W2^T + b2
//   Zb = mu + eps * sigma   (f16, padded cols are exactly 0)
// ---------------------------------------------------------------------------
__global__ __launch_bounds__(32) void mlp_kernel(
    const _Float16* __restrict__ Xin, const float* __restrict__ eps,
    const _Float16* __restrict__ W1mu, const _Float16* __restrict__ W2mu,
    const _Float16* __restrict__ W1sg, const _Float16* __restrict__ W2sg,
    const float* __restrict__ mu_b1, const float* __restrict__ mu_g,
    const float* __restrict__ mu_be, const float* __restrict__ mu_b2,
    const float* __restrict__ sg_b1, const float* __restrict__ sg_g,
    const float* __restrict__ sg_be, const float* __restrict__ sg_b2,
    _Float16* __restrict__ Zb) {
  __shared__ float    hF[16 * HID];    // pre-LN activations
  __shared__ _Float16 hH[16 * HID];    // post-GELU, f16, A-ready
  __shared__ float    muO[16 * PAD];   // mu branch output

  const size_t row0 = (size_t)blockIdx.x * 16;
  const int l = threadIdx.x, n = l & 15, kh = l >> 4;

  // A fragments of x_in (reused by both branches)
  v16h a0 = load_a_f16(Xin + row0 * PAD, PAD, 0);
  v16h a1 = load_a_f16(Xin + row0 * PAD, PAD, 32);
  v16h a2 = load_a_f16(Xin + row0 * PAD, PAD, 64);

  for (int p = 0; p < 2; ++p) {
    const _Float16* W1 = p ? W1sg : W1mu;
    const _Float16* W2 = p ? W2sg : W2mu;
    const float* b1 = p ? sg_b1 : mu_b1;
    const float* g  = p ? sg_g  : mu_g;
    const float* be = p ? sg_be : mu_be;
    const float* b2 = p ? sg_b2 : mu_b2;

    // ---- GEMM: h (16x256) = x_in @ W1^T + b1  (prefetch next N-tile) ----
    {
      v16h w0 = load_b(W1, PAD, 0, 0);
      v16h w1v = load_b(W1, PAD, 0, 32);
      v16h w2v = load_b(W1, PAD, 0, 64);
      for (int nt = 0; nt < 16; ++nt) {
        const int nn = (nt < 15) ? (nt + 1) * 16 : 0;
        v16h p0 = load_b(W1, PAD, nn, 0);
        v16h p1 = load_b(W1, PAD, nn, 32);
        v16h p2 = load_b(W1, PAD, nn, 64);
        v8f cc = zero8();
        cc = wmma_f32(a0, w0,  cc);
        cc = wmma_f32(a1, w1v, cc);
        cc = wmma_f32(a2, w2v, cc);
        const int col = nt * 16 + n;
#pragma unroll
        for (int e = 0; e < 8; ++e) hF[(e + 8 * kh) * HID + col] = cc[e] + b1[col];
        w0 = p0; w1v = p1; w2v = p2;
      }
    }
    __syncthreads();

    // ---- LayerNorm + exact GELU (2 lanes per row, 128 cols each) ----
    {
      const int row = l & 15, half = l >> 4;
      const float* hr = hF + row * HID + half * 128;
      float s = 0.f;
      for (int i = 0; i < 128; ++i) s += hr[i];
      s += __shfl_xor(s, 16, 32);
      const float mean = s * (1.f / 256.f);
      float v = 0.f;
      for (int i = 0; i < 128; ++i) { float d = hr[i] - mean; v += d * d; }
      v += __shfl_xor(v, 16, 32);
      const float rstd = rsqrtf(v * (1.f / 256.f) + 1e-5f);
      _Float16* ho = hH + row * HID + half * 128;
      const float* gg = g + half * 128;
      const float* bb = be + half * 128;
      for (int i = 0; i < 128; ++i) {
        float t = (hr[i] - mean) * rstd * gg[i] + bb[i];
        ho[i] = (_Float16)(0.5f * t * (1.f + erff(t * 0.70710678118f)));
      }
    }
    __syncthreads();

    // ---- GEMM: o (16x96) = h @ W2^T + b2  (prefetch next K-step) ----
    v8f c2[6];
#pragma unroll
    for (int t = 0; t < 6; ++t) c2[t] = zero8();
    {
      v16h ah = load_a_f16((const _Float16*)hH, HID, 0);
      v16h bw[6];
#pragma unroll
      for (int t = 0; t < 6; ++t) bw[t] = load_b(W2, HID, t * 16, 0);
      for (int ks = 0; ks < 8; ++ks) {
        const int kn = (ks < 7) ? (ks + 1) * 32 : 0;
        v16h an = load_a_f16((const _Float16*)hH, HID, kn);
        v16h bn[6];
#pragma unroll
        for (int t = 0; t < 6; ++t) bn[t] = load_b(W2, HID, t * 16, kn);
#pragma unroll
        for (int t = 0; t < 6; ++t) c2[t] = wmma_f32(ah, bw[t], c2[t]);
        ah = an;
#pragma unroll
        for (int t = 0; t < 6; ++t) bw[t] = bn[t];
      }
    }

    if (p == 0) {               // stash mu
      for (int t = 0; t < 6; ++t) {
        const int col = t * 16 + n;
        const float bv = (col < MLP_IN) ? b2[col] : 0.f;
#pragma unroll
        for (int e = 0; e < 8; ++e) muO[(e + 8 * kh) * PAD + col] = c2[t][e] + bv;
      }
    } else {                    // z = mu + eps * sigma
      for (int t = 0; t < 6; ++t) {
        const int col = t * 16 + n;
        const float bv = (col < MLP_IN) ? b2[col] : 0.f;
#pragma unroll
        for (int e = 0; e < 8; ++e) {
          const int m = e + 8 * kh;
          const float sgv = c2[t][e] + bv;
          const float ev  = (col < MLP_IN) ? eps[(row0 + m) * MLP_IN + col] : 0.f;
          const float z   = muO[m * PAD + col] + ev * sgv;
          Zb[(row0 + m) * PAD + col] = (_Float16)z;
        }
      }
    }
    __syncthreads();
  }
}

// ---------------------------------------------------------------------------
// Kernel 3: out (32768x768, f32) = Zb (f16) @ Gm^T  (sparse irfft as GEMM)
// 4 waves/block; wave = 16 rows x 128 cols (blockIdx.y selects col block).
// Prefetch next N-tile's 3 B fragments while current tile's WMMAs issue.
// ---------------------------------------------------------------------------
__global__ __launch_bounds__(128) void gemm3_kernel(const _Float16* __restrict__ Zb,
                                                    const _Float16* __restrict__ Gm,
                                                    float* __restrict__ out) {
  const int wave = threadIdx.x >> 5;
  const size_t row0 = (size_t)(blockIdx.x * 4 + wave) * 16;
  const int col0 = blockIdx.y * 128;

  v16h a0 = load_a_f16(Zb + row0 * PAD, PAD, 0);
  v16h a1 = load_a_f16(Zb + row0 * PAD, PAD, 32);
  v16h a2 = load_a_f16(Zb + row0 * PAD, PAD, 64);

  v16h b0 = load_b(Gm, PAD, col0, 0);
  v16h b1 = load_b(Gm, PAD, col0, 32);
  v16h b2 = load_b(Gm, PAD, col0, 64);

  const int l = threadIdx.x & 31, n = l & 15, kh = l >> 4;
#pragma unroll
  for (int t = 0; t < 8; ++t) {
    const int cn = col0 + ((t < 7) ? (t + 1) * 16 : 0);
    v16h n0 = load_b(Gm, PAD, cn, 0);
    v16h n1 = load_b(Gm, PAD, cn, 32);
    v16h n2 = load_b(Gm, PAD, cn, 64);

    v8f c = zero8();
    c = wmma_f32(a0, b0, c);
    c = wmma_f32(a1, b1, c);
    c = wmma_f32(a2, b2, c);
#pragma unroll
    for (int e = 0; e < 8; ++e)
      out[(row0 + e + 8 * kh) * IN_DIM + col0 + t * 16 + n] = c[e];

    b0 = n0; b1 = n1; b2 = n2;
  }
}

// ---------------------------------------------------------------------------
extern "C" void kernel_launch(void* const* d_in, const int* in_sizes, int n_in,
                              void* d_out, int out_size, void* d_ws, size_t ws_size,
                              hipStream_t stream) {
  const float* x      = (const float*)d_in[0];
  const float* eps    = (const float*)d_in[1];
  const int*   idx    = (const int*)  d_in[2];
  const float* mu_w1  = (const float*)d_in[3];
  const float* mu_b1  = (const float*)d_in[4];
  const float* mu_g   = (const float*)d_in[5];
  const float* mu_be  = (const float*)d_in[6];
  const float* mu_w2  = (const float*)d_in[7];
  const float* mu_b2  = (const float*)d_in[8];
  const float* sg_w1  = (const float*)d_in[9];
  const float* sg_b1  = (const float*)d_in[10];
  const float* sg_g   = (const float*)d_in[11];
  const float* sg_be  = (const float*)d_in[12];
  const float* sg_w2  = (const float*)d_in[13];
  const float* sg_b2  = (const float*)d_in[14];
  float* out = (float*)d_out;

  char* ws = (char*)d_ws;
  size_t off = 0;
  _Float16* FbT  = (_Float16*)(ws + off); off += (size_t)PAD * IN_DIM * 2;    // 96x768
  _Float16* Gm   = (_Float16*)(ws + off); off += (size_t)IN_DIM * PAD * 2;    // 768x96
  _Float16* W1mu = (_Float16*)(ws + off); off += (size_t)HID * PAD * 2;
  _Float16* W1sg = (_Float16*)(ws + off); off += (size_t)HID * PAD * 2;
  _Float16* W2mu = (_Float16*)(ws + off); off += (size_t)PAD * HID * 2;
  _Float16* W2sg = (_Float16*)(ws + off); off += (size_t)PAD * HID * 2;
  _Float16* Xin  = (_Float16*)(ws + off); off += (size_t)NROWS * PAD * 2;
  _Float16* Zb   = (_Float16*)(ws + off); off += (size_t)NROWS * PAD * 2;

  setup_kernel<<<64, 256, 0, stream>>>(idx, mu_w1, mu_w2, sg_w1, sg_w2,
                                       FbT, Gm, W1mu, W1sg, W2mu, W2sg);

  gemm1_kernel<<<NROWS / 64, 128, 0, stream>>>(x, FbT, Xin);

  mlp_kernel<<<NROWS / 16, 32, 0, stream>>>(Xin, eps, W1mu, W2mu, W1sg, W2sg,
                                            mu_b1, mu_g, mu_be, mu_b2,
                                            sg_b1, sg_g, sg_be, sg_b2, Zb);

  gemm3_kernel<<<dim3(NROWS / 64, IN_DIM / 128), 128, 0, stream>>>(Zb, Gm, out);
}
